// TemporalMemoryModule_10599979286560
// MI455X (gfx1250) — compile-verified, matrix operand order checked by reading, not run on hardware
//
#include <hip/hip_runtime.h>
#include <hip/hip_bf16.h>
#include <math.h>

typedef __attribute__((ext_vector_type(16))) _Float16 v16h;
typedef __attribute__((ext_vector_type(8)))  float    v8f;

#define C_DIM   256
#define M_DIM   512
#define HID     128
#define HW      4096      // 64*64
#define N_IMG   16
#define NPIX    (N_IMG * HW)   // 65536
#define TAU     0.0025f
#define EPS     1e-12f

// Per-lane K index for 16-bit WMMA A/B fragments (16x32 striping).
// lanes 0-15: VGPR0..3 hold K=0..7 (pairs), VGPR4..7 hold K=16..23
// lanes 16-31: K base shifted by 8.
__device__ __forceinline__ int kmap(int lane, int t) {
    int kb = (lane >= 16) ? 8 : 0;
    return (t < 8) ? (kb + t) : (16 + kb + (t - 8));
}

// ---------------------------------------------------------------------------
// Kernel 1: shared MLP over memory slots + L2 normalize -> mn[512][256] f32
// one block per memory row, 128 threads
// ---------------------------------------------------------------------------
__global__ __launch_bounds__(128) void mlp_kernel(
    const float* __restrict__ memory, const float* __restrict__ w1,
    const float* __restrict__ b1, const float* __restrict__ w2,
    const float* __restrict__ b2, float* __restrict__ mn)
{
    __shared__ float row[C_DIM];
    __shared__ float hid[HID];
    __shared__ float red[HID];

    const int r = blockIdx.x;
    const int t = threadIdx.x;

    row[t]       = memory[r * C_DIM + t];
    row[t + 128] = memory[r * C_DIM + t + 128];
    __syncthreads();

    float h = b1[t];
    #pragma unroll 8
    for (int c = 0; c < C_DIM; ++c)
        h = fmaf(row[c], w1[c * HID + t], h);
    hid[t] = fmaxf(h, 0.0f);
    __syncthreads();

    float o0 = b2[t], o1 = b2[t + 128];
    #pragma unroll 8
    for (int hh = 0; hh < HID; ++hh) {
        float hv = hid[hh];
        o0 = fmaf(hv, w2[hh * C_DIM + t], o0);
        o1 = fmaf(hv, w2[hh * C_DIM + t + 128], o1);
    }
    o0 = fmaxf(o0, 0.0f);
    o1 = fmaxf(o1, 0.0f);

    red[t] = o0 * o0 + o1 * o1;
    __syncthreads();
    for (int s = 64; s > 0; s >>= 1) {
        if (t < s) red[t] += red[t + s];
        __syncthreads();
    }
    float inv = 1.0f / fmaxf(sqrtf(red[0]), EPS);

    mn[r * C_DIM + t]       = o0 * inv;
    mn[r * C_DIM + t + 128] = o1 * inv;
}

// ---------------------------------------------------------------------------
// Kernel 2: pack mem_norm into f16 WMMA B-fragment layouts (lane-major).
//  memTB: B for GEMM1 (att = Xn @ Mnᵀ): B[k=c][n=m]
//  memB2: B for GEMM2 (out = att @ Mn):  B[k=m][n=c]
// ---------------------------------------------------------------------------
__global__ __launch_bounds__(256) void pack_kernel(
    const float* __restrict__ mn, _Float16* __restrict__ memTB,
    _Float16* __restrict__ memB2)
{
    const int idx = blockIdx.x * blockDim.x + threadIdx.x; // 0..262143
    const int t    = idx & 15;
    const int lane = (idx >> 4) & 31;
    const int frag = idx >> 9;
    const int L = M_DIM * C_DIM; // 131072 per layout

    if (idx < L) {
        int j = frag >> 3, s = frag & 7;
        int m = 16 * j + (lane & 15);
        int c = 32 * s + kmap(lane, t);
        memTB[idx] = (_Float16)mn[m * C_DIM + c];
    } else {
        int f2 = frag - (L >> 9);
        int j2 = f2 >> 4, s2 = f2 & 15;
        int c = 16 * j2 + (lane & 15);
        int m = 32 * s2 + kmap(lane, t);
        memB2[idx - L] = (_Float16)mn[m * C_DIM + c];
    }
}

// ---------------------------------------------------------------------------
// Kernel 3: fused pipeline. 1 wave / block, 32 pixels / wave (2 row-groups).
// GEMM1+softmax per group into a reusable f32 LDS slab; final weights staged
// in LDS as f16 A-fragments; GEMM2 shares each B fragment across both groups.
// ---------------------------------------------------------------------------
__global__ __launch_bounds__(32) void fused_kernel(
    const float* __restrict__ x, const _Float16* __restrict__ memTB,
    const _Float16* __restrict__ memB2, float* __restrict__ out,
    float* __restrict__ attmap)
{
    __shared__ float    f32buf[16 * M_DIM];            // 32 KB, reused per group
    __shared__ _Float16 f16st[2][16 * 32 * 16];        // 2 x 16 KB staged A frags

    const int lane = threadIdx.x & 31;
    const int p0      = blockIdx.x * 32;
    const int n       = p0 >> 12;        // / HW
    const int hw_base = p0 & (HW - 1);
    const int rr      = lane & 15;       // row within 16-row group
    const int half    = lane >> 4;       // 0 or 1
    const int rbase   = half * 8;

    for (int g = 0; g < 2; ++g) {
        const int pix0 = hw_base + g * 16;               // group's first pixel (hw)
        const int xoff = n * (C_DIM * HW) + pix0;

        // ---- per-pixel sum of squares (lane pairs split channels)
        float ss = 0.0f;
        for (int c = half; c < C_DIM; c += 2) {
            float v = x[xoff + c * HW + rr];
            ss = fmaf(v, v, ss);
        }
        ss += __shfl_xor(ss, 16, 32);
        const float invn = 1.0f / fmaxf(sqrtf(ss), EPS);

        // ---- normalized f16 A fragments (8 K-steps of 32)
        v16h a[8];
        #pragma unroll
        for (int s = 0; s < 8; ++s) {
            #pragma unroll
            for (int t = 0; t < 16; ++t) {
                int c = 32 * s + kmap(lane, t);
                a[s][t] = (_Float16)(x[xoff + c * HW + rr] * invn);
            }
        }

        // ---- GEMM1: logits[16 x 512] = Xn @ Mnᵀ  -> f32buf
        for (int j = 0; j < 32; ++j) {
            v8f acc = {};
            #pragma unroll
            for (int s = 0; s < 8; ++s) {
                const v16h b = *(const v16h*)(memTB + (((j * 8 + s) * 32 + lane) << 4));
                acc = __builtin_amdgcn_wmma_f32_16x16x32_f16(
                    false, a[s], false, b, (short)0, acc, false, false);
            }
            const int col = 16 * j + rr;
            #pragma unroll
            for (int e = 0; e < 8; ++e)
                f32buf[(rbase + e) * M_DIM + col] = acc[e];
        }

        // ---- softmax + hardshrink + L1 renorm (2 lanes/row, 256 cols each)
        float* rowp = f32buf + rr * M_DIM + half * 256;

        float mx = -3.4e38f;
        for (int i = 0; i < 256; ++i) mx = fmaxf(mx, rowp[i]);
        mx = fmaxf(mx, __shfl_xor(mx, 16, 32));

        float s1 = 0.0f;
        for (int i = 0; i < 256; ++i) {
            float e = __expf(rowp[i] - mx);
            rowp[i] = e;
            s1 += e;
        }
        s1 += __shfl_xor(s1, 16, 32);
        const float inv1 = 1.0f / s1;

        float s2sum = 0.0f;
        for (int i = 0; i < 256; ++i) {
            float v = fmaxf(rowp[i] * inv1 - TAU, 0.0f);
            rowp[i] = v;
            s2sum += v;
        }
        s2sum += __shfl_xor(s2sum, 16, 32);
        const float inv2 = 1.0f / fmaxf(s2sum, EPS);

        // ---- finalize: write att_map + stage f16 A-fragments for GEMM2
        const int amap_off = n * (M_DIM * HW) + pix0 + rr;
        for (int s2g = 0; s2g < 8; ++s2g) {
            const int s2 = half * 8 + s2g;
            _Float16* blo = &f16st[g][((s2 * 32) + rr) * 16];       // target lane rr
            _Float16* bhi = &f16st[g][((s2 * 32) + rr + 16) * 16];  // target lane rr+16
            #pragma unroll
            for (int km = 0; km < 32; ++km) {
                const int i = s2g * 32 + km;
                float v = rowp[i] * inv2;
                attmap[amap_off + (half * 256 + i) * HW] = v;
                _Float16 hv = (_Float16)v;
                if      (km < 8)  blo[km]      = hv;   // K 0..7   -> t 0..7
                else if (km < 16) bhi[km - 8]  = hv;   // K 8..15  -> t 0..7 (hi lane)
                else if (km < 24) blo[km - 8]  = hv;   // K 16..23 -> t 8..15
                else              bhi[km - 16] = hv;   // K 24..31 -> t 8..15 (hi lane)
            }
        }
    }

    // ---- GEMM2: out[32 x 256] = att @ Mn ; each B fragment feeds both groups
    const int ooff = n * (C_DIM * HW) + hw_base;
    for (int j2 = 0; j2 < 16; ++j2) {
        v8f acc0 = {};
        v8f acc1 = {};
        for (int s2 = 0; s2 < 16; ++s2) {
            const v16h a20 = *(const v16h*)(&f16st[0][((s2 * 32) + lane) << 4]);
            const v16h a21 = *(const v16h*)(&f16st[1][((s2 * 32) + lane) << 4]);
            const v16h b   = *(const v16h*)(memB2 + (((j2 * 16 + s2) * 32 + lane) << 4));
            acc0 = __builtin_amdgcn_wmma_f32_16x16x32_f16(
                false, a20, false, b, (short)0, acc0, false, false);
            acc1 = __builtin_amdgcn_wmma_f32_16x16x32_f16(
                false, a21, false, b, (short)0, acc1, false, false);
        }
        const int c = 16 * j2 + rr;
        #pragma unroll
        for (int e = 0; e < 8; ++e) {
            out[ooff + c * HW + rbase + e]      = acc0[e];   // group 0 pixels
            out[ooff + c * HW + 16 + rbase + e] = acc1[e];   // group 1 pixels
        }
    }
}

// ---------------------------------------------------------------------------
extern "C" void kernel_launch(void* const* d_in, const int* in_sizes, int n_in,
                              void* d_out, int out_size, void* d_ws, size_t ws_size,
                              hipStream_t stream) {
    const float* x      = (const float*)d_in[0];
    const float* memory = (const float*)d_in[1];
    const float* w1     = (const float*)d_in[2];
    const float* b1     = (const float*)d_in[3];
    const float* w2     = (const float*)d_in[4];
    const float* b2     = (const float*)d_in[5];

    float* out    = (float*)d_out;                      // [16,256,64,64]
    float* attmap = out + (size_t)NPIX * C_DIM;         // [16,512,64,64]

    float*    mn    = (float*)d_ws;                               // 512KB
    _Float16* memTB = (_Float16*)((char*)d_ws + 524288);          // 256KB
    _Float16* memB2 = memTB + M_DIM * C_DIM;                      // 256KB

    mlp_kernel<<<M_DIM, 128, 0, stream>>>(memory, w1, b1, w2, b2, mn);
    pack_kernel<<<(2 * M_DIM * C_DIM) / 256, 256, 0, stream>>>(mn, memTB, memB2);
    fused_kernel<<<NPIX / 32, 32, 0, stream>>>(x, memTB, memB2, out, attmap);
}
